// Qwen2VisionTransformerPretrainedModelMock_52432960750180
// MI455X (gfx1250) — compile-verified, matrix operand order checked by reading, not run on hardware
//
#include <hip/hip_runtime.h>
#include <hip/hip_bf16.h>
#include <math.h>

typedef __bf16 bf16;
typedef __attribute__((ext_vector_type(16))) __bf16 v16bf;
typedef __attribute__((ext_vector_type(8)))  __bf16 v8bf;
typedef __attribute__((ext_vector_type(4)))  __bf16 v4bf;
typedef __attribute__((ext_vector_type(8)))  float  v8f;

#define SEQ      2048
#define EMBED    1280
#define HEADS    16
#define HD       80
#define HDP      96        // HD padded to 3*32 for WMMA K-chunks
#define MLPDIM   5120
#define HIDDEN   3584
#define PATCH_IN 1176
#define SEGLEN   1024
#define DEPTH    2

#define BM 128
#define BN 128
#define BK 32

// ---------------------------------------------------------------------------
// Generic GEMM: C[M,N] = act(A[M,K](f32) * B[K,N](f32) + bias) + residual
// f32 operands converted to bf16 while staging into LDS; math is
// v_wmma_f32_16x16x32_bf16 with f32 accumulation.
// Block tile 128x128, BK=32, 256 threads = 8 waves (4x2), each wave 32x64
// (2x4 accumulators, 8 WMMA / k-step). DOUBLE-BUFFERED LDS: one barrier per
// k-step; next slab is prefetched to registers and committed to the idle
// buffer while WMMAs consume the current one.
// Requires M%128==0 and N%128==0 (true for every call here).
// K: branch-free pipelined main loop (K&~31) + one guarded tail iteration.
// act: 0=none, 1=silu, 2=gelu(erf)
// ---------------------------------------------------------------------------
__global__ __launch_bounds__(256)
void gemm_kernel(const float* __restrict__ A, int lda,
                 const float* __restrict__ B, int ldb,
                 const float* __restrict__ bias,
                 const float* __restrict__ residual,
                 float* __restrict__ C, int ldc,
                 int K, int act)
{
    __shared__ __align__(32) bf16 As[2][BM * BK];   // [m][k]            2 x 8 KB
    __shared__ __align__(32) bf16 Bs[2][BN * BK];   // [n][k] transposed 2 x 8 KB

    const int tid  = threadIdx.x;
    const int lane = tid & 31;
    const int hi   = (lane >> 4) & 1;
    const int ln16 = lane & 15;
    const int w    = tid >> 5;          // 0..7
    const int wm   = w >> 1;            // 0..3 : M = 4 x 32
    const int wn   = w & 1;             // 0..1 : N = 2 x 64
    const int bm   = blockIdx.y * BM;
    const int bn   = blockIdx.x * BN;

    // A staging: thread -> row (tid>>1), 16 k-elems at (tid&1)*16
    const int arow = tid >> 1;
    const int acg  = (tid & 1) * 16;
    // B staging: thread -> n-pair 2*(tid&63), 8 k-elems at (tid>>6)*8
    const int bn2  = (tid & 63) * 2;
    const int bk8  = (tid >> 6) * 8;

    v8f acc[2][4] = {};

    float4 a4[4];   // prefetch registers
    float2 b2[8];

    auto stage = [&](int k0) {   // bulk global loads (no bounds checks)
        const float4* ap = (const float4*)(A + (size_t)(bm + arow) * lda + k0 + acg);
        #pragma unroll
        for (int j = 0; j < 4; ++j) a4[j] = ap[j];
        const float* bp = B + (size_t)(k0 + bk8) * ldb + bn + bn2;
        #pragma unroll
        for (int j = 0; j < 8; ++j) b2[j] = *(const float2*)(bp + (size_t)j * ldb);
    };

    auto commit = [&](int buf) {  // convert + vectorized LDS stores
        #pragma unroll
        for (int j = 0; j < 4; ++j) {
            v4bf t = {(bf16)a4[j].x, (bf16)a4[j].y, (bf16)a4[j].z, (bf16)a4[j].w};
            *(v4bf*)&As[buf][arow * BK + acg + j * 4] = t;
        }
        v8bf t0, t1;
        #pragma unroll
        for (int j = 0; j < 8; ++j) { t0[j] = (bf16)b2[j].x; t1[j] = (bf16)b2[j].y; }
        *(v8bf*)&Bs[buf][(bn2 + 0) * BK + bk8] = t0;
        *(v8bf*)&Bs[buf][(bn2 + 1) * BK + bk8] = t1;
    };

    auto mma_step = [&](int buf) {
        v16bf a[2], b[4];
        #pragma unroll
        for (int t = 0; t < 2; ++t)
            a[t] = *(const v16bf*)&As[buf][(wm * 32 + t * 16 + ln16) * BK + hi * 16];
        #pragma unroll
        for (int t = 0; t < 4; ++t)
            b[t] = *(const v16bf*)&Bs[buf][(wn * 64 + t * 16 + ln16) * BK + hi * 16];
        #pragma unroll
        for (int tm = 0; tm < 2; ++tm)
            #pragma unroll
            for (int tn = 0; tn < 4; ++tn)
                acc[tm][tn] = __builtin_amdgcn_wmma_f32_16x16x32_bf16(
                    false, a[tm], false, b[tn], (short)0, acc[tm][tn], false, false);
    };

    const int nsteps = (K & ~(BK - 1)) / BK;

    if (nsteps > 0) {
        stage(0);
        commit(0);
        __syncthreads();
        for (int s = 0; s < nsteps; ++s) {
            const bool more = (s + 1 < nsteps);
            if (more) stage((s + 1) * BK);     // global prefetch overlaps WMMAs
            mma_step(s & 1);
            if (more) commit((s + 1) & 1);     // write idle buffer, no conflict
            __syncthreads();                   // single barrier per k-step
        }
    }

    if (K & (BK - 1)) {   // guarded tail (patch GEMM only: K = 1176)
        const int k0  = nsteps * BK;
        const int buf = nsteps & 1;            // idle buffer after the loop
        #pragma unroll
        for (int j = 0; j < 16; ++j) {
            const int kk = k0 + acg + j;
            const float v = (kk < K) ? A[(size_t)(bm + arow) * lda + kk] : 0.0f;
            As[buf][arow * BK + acg + j] = (bf16)v;
        }
        #pragma unroll
        for (int j = 0; j < 8; ++j) {
            const int kk = k0 + bk8 + j;
            float v0 = 0.0f, v1 = 0.0f;
            if (kk < K) {
                v0 = B[(size_t)kk * ldb + bn + bn2 + 0];
                v1 = B[(size_t)kk * ldb + bn + bn2 + 1];
            }
            Bs[buf][(bn2 + 0) * BK + bk8 + j] = (bf16)v0;
            Bs[buf][(bn2 + 1) * BK + bk8 + j] = (bf16)v1;
        }
        __syncthreads();
        mma_step(buf);
    }

    // epilogue: C layout -> (VGPR r, lane) = row r + hi*8, col ln16
    #pragma unroll
    for (int tm = 0; tm < 2; ++tm) {
        #pragma unroll
        for (int tn = 0; tn < 4; ++tn) {
            #pragma unroll
            for (int r = 0; r < 8; ++r) {
                const int row = bm + wm * 32 + tm * 16 + r + hi * 8;
                const int col = bn + wn * 64 + tn * 16 + ln16;
                float v = acc[tm][tn][r];
                if (bias) v += bias[col];
                if (act == 1)      v = v / (1.0f + __expf(-v));                              // silu
                else if (act == 2) v = 0.5f * v * (1.0f + erff(v * 0.70710678118654752f));   // gelu exact
                if (residual) v += residual[(size_t)row * ldc + col];
                C[(size_t)row * ldc + col] = v;
            }
        }
    }
}

// ---------------------------------------------------------------------------
// Row LayerNorm (eps=1e-6), one 256-thread block per row.
// ---------------------------------------------------------------------------
__global__ __launch_bounds__(256)
void ln_kernel(const float* __restrict__ x, const float* __restrict__ w,
               const float* __restrict__ b, float* __restrict__ out, int cols)
{
    __shared__ float red[256];
    const int row = blockIdx.x;
    const float* xr = x + (size_t)row * cols;

    float s = 0.0f;
    for (int c = threadIdx.x; c < cols; c += 256) s += xr[c];
    red[threadIdx.x] = s; __syncthreads();
    for (int st = 128; st > 0; st >>= 1) {
        if (threadIdx.x < st) red[threadIdx.x] += red[threadIdx.x + st];
        __syncthreads();
    }
    const float mu = red[0] / cols;
    __syncthreads();

    float vs = 0.0f;
    for (int c = threadIdx.x; c < cols; c += 256) { float d = xr[c] - mu; vs += d * d; }
    red[threadIdx.x] = vs; __syncthreads();
    for (int st = 128; st > 0; st >>= 1) {
        if (threadIdx.x < st) red[threadIdx.x] += red[threadIdx.x + st];
        __syncthreads();
    }
    const float rstd = rsqrtf(red[0] / cols + 1e-6f);
    __syncthreads();

    float* orow = out + (size_t)row * cols;
    for (int c = threadIdx.x; c < cols; c += 256)
        orow[c] = (xr[c] - mu) * rstd * w[c] + b[c];
}

// ---------------------------------------------------------------------------
// RoPE + repack: qkv f32 (SEQ, 3, HEADS, 80) ->
//   Qp,Kp bf16 (HEADS, SEQ, 96)  [rotary applied, d=80..95 zero]
//   Vt    bf16 (HEADS, 96, SEQ)  [transposed so P*V B-frags are contiguous]
// ---------------------------------------------------------------------------
__global__ __launch_bounds__(256)
void rope_kernel(const float* __restrict__ qkv, const int* __restrict__ pos,
                 bf16* __restrict__ Qp, bf16* __restrict__ Kp, bf16* __restrict__ Vt)
{
    const int s = blockIdx.x;
    const float hp = (float)pos[s * 2 + 0];
    const float wp = (float)pos[s * 2 + 1];
    const float* base = qkv + (size_t)s * 3 * EMBED;

    for (int idx = threadIdx.x; idx < HEADS * HDP; idx += 256) {
        const int h = idx / HDP, d = idx % HDP;
        const size_t qk_off = ((size_t)h * SEQ + s) * HDP + d;
        const size_t vt_off = ((size_t)(h * HDP + d)) * SEQ + s;
        if (d < HD) {
            const int dd = d % 40, j = dd % 20;
            const float p   = (dd < 20) ? hp : wp;
            const float ang = p * __powf(10000.0f, -(float)j / 20.0f);
            const float cs = __cosf(ang), sn = __sinf(ang);
            const float qv = base[h * HD + d];
            const float kv = base[EMBED + h * HD + d];
            const float vv = base[2 * EMBED + h * HD + d];
            const float qr = (d < 40) ? -base[h * HD + d + 40] : base[h * HD + d - 40];
            const float kr = (d < 40) ? -base[EMBED + h * HD + d + 40] : base[EMBED + h * HD + d - 40];
            Qp[qk_off] = (bf16)(qv * cs + qr * sn);
            Kp[qk_off] = (bf16)(kv * cs + kr * sn);
            Vt[vt_off] = (bf16)vv;
        } else {
            Qp[qk_off] = (bf16)0.0f;
            Kp[qk_off] = (bf16)0.0f;
            Vt[vt_off] = (bf16)0.0f;
        }
    }
}

// ---------------------------------------------------------------------------
// Flash-style attention, one wave per (16-row q-tile, head).
// Keys restricted to the q-tile's 1024-token segment (block-diagonal mask).
// S = Q*K^T via 6 WMMAs per 32-key step (HD padded to 96 = 3 K-chunks),
// online softmax via 16-lane shuffle reductions, P restaged through LDS
// (C-layout -> A-layout), O += P*V via 5 WMMAs (d = 5 tiles of 16).
// ---------------------------------------------------------------------------
__global__ __launch_bounds__(32)
void attn_kernel(const bf16* __restrict__ Qp, const bf16* __restrict__ Kp,
                 const bf16* __restrict__ Vt, float* __restrict__ O)
{
    __shared__ __align__(32) bf16 Pl[16 * 32];

    const int lane = threadIdx.x & 31;
    const int hi   = lane >> 4;
    const int ln16 = lane & 15;
    const int head = blockIdx.y;
    const int q0   = blockIdx.x * 16;
    const int seg0 = (q0 / SEGLEN) * SEGLEN;

    const bf16* Qh = Qp + (size_t)head * SEQ * HDP;
    const bf16* Kh = Kp + (size_t)head * SEQ * HDP;
    const bf16* Vh = Vt + (size_t)head * HDP * SEQ;

    v16bf aq[3];
    #pragma unroll
    for (int c = 0; c < 3; ++c)
        aq[c] = *(const v16bf*)&Qh[(size_t)(q0 + ln16) * HDP + c * 32 + hi * 16];

    float m[8], l[8];
    #pragma unroll
    for (int r = 0; r < 8; ++r) { m[r] = -INFINITY; l[r] = 0.0f; }
    v8f acc[5] = {};

    const float scale = 0.11180339887498949f;  // 1/sqrt(80)

    for (int kb = 0; kb < SEGLEN; kb += 32) {
        const int kbase = seg0 + kb;
        v8f s0 = {}, s1 = {};
        #pragma unroll
        for (int c = 0; c < 3; ++c) {
            v16bf bk0 = *(const v16bf*)&Kh[(size_t)(kbase + ln16) * HDP + c * 32 + hi * 16];
            v16bf bk1 = *(const v16bf*)&Kh[(size_t)(kbase + 16 + ln16) * HDP + c * 32 + hi * 16];
            s0 = __builtin_amdgcn_wmma_f32_16x16x32_bf16(false, aq[c], false, bk0, (short)0, s0, false, false);
            s1 = __builtin_amdgcn_wmma_f32_16x16x32_bf16(false, aq[c], false, bk1, (short)0, s1, false, false);
        }
        // online softmax; per-lane VGPR r = row (r + hi*8), lanes of a 16-group = columns
        #pragma unroll
        for (int r = 0; r < 8; ++r) {
            const float v0 = s0[r] * scale, v1 = s1[r] * scale;
            float mx = fmaxf(v0, v1);
            #pragma unroll
            for (int off = 8; off > 0; off >>= 1)
                mx = fmaxf(mx, __shfl_xor(mx, off, 16));
            const float mnew  = fmaxf(m[r], mx);
            const float alpha = __expf(m[r] - mnew);
            const float p0 = __expf(v0 - mnew);
            const float p1 = __expf(v1 - mnew);
            float rs = p0 + p1;
            #pragma unroll
            for (int off = 8; off > 0; off >>= 1)
                rs += __shfl_xor(rs, off, 16);
            l[r] = l[r] * alpha + rs;
            m[r] = mnew;
            #pragma unroll
            for (int n = 0; n < 5; ++n) acc[n][r] *= alpha;
            Pl[(r + hi * 8) * 32 + ln16]      = (bf16)p0;
            Pl[(r + hi * 8) * 32 + 16 + ln16] = (bf16)p1;
        }
        __syncthreads();
        v16bf ap = *(const v16bf*)&Pl[ln16 * 32 + hi * 16];
        #pragma unroll
        for (int n = 0; n < 5; ++n) {
            v16bf bv = *(const v16bf*)&Vh[(size_t)(n * 16 + ln16) * SEQ + kbase + hi * 16];
            acc[n] = __builtin_amdgcn_wmma_f32_16x16x32_bf16(false, ap, false, bv, (short)0, acc[n], false, false);
        }
        __syncthreads();
    }

    #pragma unroll
    for (int n = 0; n < 5; ++n) {
        #pragma unroll
        for (int r = 0; r < 8; ++r) {
            const int row = q0 + r + hi * 8;
            const int col = head * HD + n * 16 + ln16;
            O[(size_t)row * EMBED + col] = acc[n][r] / l[r];
        }
    }
}

// ---------------------------------------------------------------------------
extern "C" void kernel_launch(void* const* d_in, const int* in_sizes, int n_in,
                              void* d_out, int out_size, void* d_ws, size_t ws_size,
                              hipStream_t stream)
{
    (void)in_sizes; (void)n_in; (void)out_size; (void)ws_size;

    const float* hidden   = (const float*)d_in[0];
    const int*   pos_ids  = (const int*)d_in[1];
    /* d_in[2] cu_seqlens: segment structure is fixed (2 x 1024) */
    const float* patch_w  = (const float*)d_in[3];
    const float* ln1_w    = (const float*)d_in[4];
    const float* ln1_b    = (const float*)d_in[5];
    const float* ln2_w    = (const float*)d_in[6];
    const float* ln2_b    = (const float*)d_in[7];
    const float* qkv_w    = (const float*)d_in[8];
    const float* qkv_b    = (const float*)d_in[9];
    const float* proj_w   = (const float*)d_in[10];
    const float* proj_b   = (const float*)d_in[11];
    const float* fc1_w    = (const float*)d_in[12];
    const float* fc1_b    = (const float*)d_in[13];
    const float* fc2_w    = (const float*)d_in[14];
    const float* fc2_b    = (const float*)d_in[15];
    const float* mrg_ln_w = (const float*)d_in[16];
    const float* mrg_ln_b = (const float*)d_in[17];
    const float* mrg_w1   = (const float*)d_in[18];
    const float* mrg_b1   = (const float*)d_in[19];
    const float* mrg_w2   = (const float*)d_in[20];
    const float* mrg_b2   = (const float*)d_in[21];
    float* out = (float*)d_out;

    char* ws = (char*)d_ws;
    size_t off = 0;
    auto alloc = [&](size_t bytes) -> char* {
        char* p = ws + off;
        off += (bytes + 255) & ~((size_t)255);
        return p;
    };
    float* x    = (float*)alloc((size_t)SEQ * EMBED * 4);
    float* h    = (float*)alloc((size_t)SEQ * EMBED * 4);
    float* qkvb = (float*)alloc((size_t)SEQ * 3 * EMBED * 4);
    float* ob   = (float*)alloc((size_t)SEQ * EMBED * 4);
    float* mlp  = (float*)alloc((size_t)SEQ * MLPDIM * 4);
    float* y1   = (float*)alloc((size_t)(SEQ / 4) * HIDDEN * 4);
    bf16*  Qp   = (bf16*)alloc((size_t)HEADS * SEQ * HDP * 2);
    bf16*  Kp   = (bf16*)alloc((size_t)HEADS * SEQ * HDP * 2);
    bf16*  Vt   = (bf16*)alloc((size_t)HEADS * HDP * SEQ * 2);

    const dim3 blk(256);
    auto gemm = [&](const float* A, int lda, const float* B, int ldb,
                    const float* bias, const float* res, float* C,
                    int M, int N, int K, int act) {
        dim3 grid(N / BN, M / BM);
        gemm_kernel<<<grid, blk, 0, stream>>>(A, lda, B, ldb, bias, res, C, N, K, act);
    };

    // patch embed: x = hidden @ patch_w   (K=1176 -> 36 pipelined k-steps + tail)
    gemm(hidden, PATCH_IN, patch_w, EMBED, nullptr, nullptr, x, SEQ, EMBED, PATCH_IN, 0);

    for (int i = 0; i < DEPTH; ++i) {
        ln_kernel<<<SEQ, blk, 0, stream>>>(x, ln1_w + i * EMBED, ln1_b + i * EMBED, h, EMBED);
        gemm(h, EMBED, qkv_w + (size_t)i * EMBED * 3 * EMBED, 3 * EMBED,
             qkv_b + i * 3 * EMBED, nullptr, qkvb, SEQ, 3 * EMBED, EMBED, 0);
        rope_kernel<<<SEQ, blk, 0, stream>>>(qkvb, pos_ids, Qp, Kp, Vt);
        attn_kernel<<<dim3(SEQ / 16, HEADS), dim3(32), 0, stream>>>(Qp, Kp, Vt, ob);
        // x = x + ob @ proj_w + proj_b   (in-place residual: same thread reads then writes)
        gemm(ob, EMBED, proj_w + (size_t)i * EMBED * EMBED, EMBED,
             proj_b + i * EMBED, x, x, SEQ, EMBED, EMBED, 0);
        ln_kernel<<<SEQ, blk, 0, stream>>>(x, ln2_w + i * EMBED, ln2_b + i * EMBED, h, EMBED);
        gemm(h, EMBED, fc1_w + (size_t)i * EMBED * MLPDIM, MLPDIM,
             fc1_b + i * MLPDIM, nullptr, mlp, SEQ, MLPDIM, EMBED, 1 /*silu*/);
        gemm(mlp, MLPDIM, fc2_w + (size_t)i * MLPDIM * EMBED, EMBED,
             fc2_b + i * EMBED, x, x, SEQ, EMBED, MLPDIM, 0);
    }

    // merger: LN then reshape (SEQ,1280)->(512,5120) is a no-op on contiguous memory
    ln_kernel<<<SEQ, blk, 0, stream>>>(x, mrg_ln_w, mrg_ln_b, h, EMBED);
    gemm(h, 4 * EMBED, mrg_w1, HIDDEN, mrg_b1, nullptr, y1,
         SEQ / 4, HIDDEN, 4 * EMBED, 2 /*gelu*/);
    gemm(y1, HIDDEN, mrg_w2, HIDDEN, mrg_b2, nullptr, out,
         SEQ / 4, HIDDEN, HIDDEN, 0);
}
